// RecurrentNetwork_91311004713168
// MI455X (gfx1250) — matrix operand order repaired and checked by reading, not compiled, not run on hardware
//
#include <hip/hip_runtime.h>
#include <hip/hip_bf16.h>

typedef __bf16 bf16;
typedef __attribute__((ext_vector_type(16))) __bf16 v16bf;
typedef __attribute__((ext_vector_type(8)))  __bf16 v8bf;
typedef __attribute__((ext_vector_type(8)))  float  v8f;

#define HIDDEN 512
#define GATES  1536
#define BATCH  128
#define TLEN   512
#define EMBDIM 300
#define EMBPAD 320
#define NOUT   5

union FragU { v16bf v; v8bf h[2]; };

// A fragment: 16x32 bf16, row-major source [M, ld].
// ISA layout: lanes 0-15 = rows M=0..15 holding K k..k+7 and k+16..k+23;
//             lanes 16-31 = rows M=0..15 holding K k+8..k+15 and k+24..k+31.
__device__ __forceinline__ v16bf load_a_frag(const bf16* __restrict__ base, int ld, int k) {
  const int lane = threadIdx.x & 31;
  const int row  = lane & 15;
  const int koff = (lane >> 4) << 3;   // 0 or 8
  const bf16* p = base + (size_t)row * ld + k + koff;
  FragU f;
  f.h[0] = *reinterpret_cast<const v8bf*>(p);        // K k+koff   .. +7
  f.h[1] = *reinterpret_cast<const v8bf*>(p + 16);   // K k+koff+16.. +23
  return f.v;
}

// B fragment: 32x16 bf16 (K x N). Source is W stored [N, K] row-major, so
// lane = column N holds contiguous K. ISA layout: lanes 0-15 hold K k..k+7 and k+16..k+23,
// lanes 16-31 hold K k+8..k+15 and k+24..k+31.
__device__ __forceinline__ v16bf load_b_frag(const bf16* __restrict__ base, int ld, int k) {
  const int lane = threadIdx.x & 31;
  const int col  = lane & 15;
  const int koff = (lane >> 4) << 3;   // 0 or 8
  const bf16* p = base + (size_t)col * ld + k + koff;
  FragU f;
  f.h[0] = *reinterpret_cast<const v8bf*>(p);
  f.h[1] = *reinterpret_cast<const v8bf*>(p + 16);
  return f.v;
}

__device__ __forceinline__ v8f wmma_bf16(v16bf a, v16bf b, v8f c) {
  return __builtin_amdgcn_wmma_f32_16x16x32_bf16(false, a, false, b, (short)0, c, false, false);
}

// ---------------- init: zero the h-state buffers ----------------
__global__ void init_zero_kernel(bf16* __restrict__ zero_bf, bf16* __restrict__ h1bf0,
                                 float* __restrict__ h0f0, float* __restrict__ h1f0) {
  const int i = blockIdx.x * blockDim.x + threadIdx.x;
  if (i < BATCH * HIDDEN) {
    zero_bf[i] = (bf16)0.f;
    h1bf0[i]   = (bf16)0.f;
    h0f0[i]    = 0.f;
    h1f0[i]    = 0.f;
  }
}

// ---------------- embedding gather -> time-major bf16, K padded to 320 ----------------
__global__ void embed_gather_kernel(const int* __restrict__ x, const float* __restrict__ emb,
                                    bf16* __restrict__ A0) {
  const int row = blockIdx.x;          // row = t*128 + b (time-major)
  const int t = row >> 7;
  const int b = row & 127;
  const int col = threadIdx.x;         // 0..319
  const int tok = x[b * TLEN + t];
  const float v = (col < EMBDIM) ? emb[(size_t)tok * EMBDIM + col] : 0.f;
  A0[(size_t)row * EMBPAD + col] = (bf16)v;
}

// ---------------- f32 -> bf16 weight convert with K padding ----------------
__global__ void cvt_w_kernel(const float* __restrict__ W, bf16* __restrict__ Wb,
                             int kin, int kout) {
  const int n = blockIdx.x;
  const int c = threadIdx.x;
  const float v = (c < kin) ? W[(size_t)n * kin + c] : 0.f;
  Wb[(size_t)n * kout + c] = (bf16)v;
}

// ---------------- one GRU timestep (shared by both layers) ----------------
// gates = [x_t @ Wih^T + b_ih] and [h @ Whh^T + b_hh]; r/z accumulators fused
// across both GEMM phases, n kept split (reference multiplies r only into the
// hidden-side pre-activation).
__global__ __launch_bounds__(256)
void gru_step_kernel(const bf16* __restrict__ Ax, int kx,
                     const bf16* __restrict__ Wih,   // [1536, kx] bf16
                     const bf16* __restrict__ Ah,    // [128, 512] bf16 (prev h)
                     const bf16* __restrict__ Whh,   // [1536, 512] bf16
                     const float* __restrict__ b_ih,
                     const float* __restrict__ b_hh,
                     const float* __restrict__ h_old,  // [128,512] f32
                     float* __restrict__ h_new,        // [128,512] f32
                     bf16* __restrict__ h_out) {       // [128,512] bf16
  const int gw    = blockIdx.x * 8 + (threadIdx.x >> 5);  // 256 waves total
  const int mbase = (gw >> 5) << 4;    // 8 M-tiles of 16 rows
  const int jbase = (gw & 31) << 4;    // 32 j-tiles of 16 hidden columns
  const int lane  = threadIdx.x & 31;

  v8f acc_r = {}, acc_z = {}, acc_xn = {}, acc_hn = {};

  // ---- input phase: x_t @ Wih^T ----
  const bf16* Axm = Ax + (size_t)mbase * kx;
  const bf16* Wr  = Wih + (size_t)(jbase) * kx;
  const bf16* Wz  = Wih + (size_t)(HIDDEN + jbase) * kx;
  const bf16* Wn  = Wih + (size_t)(2 * HIDDEN + jbase) * kx;
  for (int k = 0; k < kx; k += 32) {
    const v16bf a = load_a_frag(Axm, kx, k);
    acc_r  = wmma_bf16(a, load_b_frag(Wr, kx, k), acc_r);
    acc_z  = wmma_bf16(a, load_b_frag(Wz, kx, k), acc_z);
    acc_xn = wmma_bf16(a, load_b_frag(Wn, kx, k), acc_xn);
  }

  // ---- hidden phase: h @ Whh^T ----
  const bf16* Ahm = Ah + (size_t)mbase * HIDDEN;
  const bf16* Hr  = Whh + (size_t)(jbase) * HIDDEN;
  const bf16* Hz  = Whh + (size_t)(HIDDEN + jbase) * HIDDEN;
  const bf16* Hn  = Whh + (size_t)(2 * HIDDEN + jbase) * HIDDEN;
  for (int k = 0; k < HIDDEN; k += 32) {
    const v16bf a = load_a_frag(Ahm, HIDDEN, k);
    acc_r  = wmma_bf16(a, load_b_frag(Hr, HIDDEN, k), acc_r);
    acc_z  = wmma_bf16(a, load_b_frag(Hz, HIDDEN, k), acc_z);
    acc_hn = wmma_bf16(a, load_b_frag(Hn, HIDDEN, k), acc_hn);
  }

  // ---- gate nonlinearity + state update ----
  // C/D layout: VGPR r, lanes 0-15 -> M=r, lanes 16-31 -> M=r+8; N = lane&15.
  const int j = jbase + (lane & 15);
  const float bir = b_ih[j], biz = b_ih[HIDDEN + j], bin = b_ih[2 * HIDDEN + j];
  const float bhr = b_hh[j], bhz = b_hh[HIDDEN + j], bhn = b_hh[2 * HIDDEN + j];
  const int mhalf = (lane >> 4) << 3;

#pragma unroll
  for (int rI = 0; rI < 8; ++rI) {
    const int m = mbase + mhalf + rI;
    const float hprev = h_old[(size_t)m * HIDDEN + j];
    const float rg = 1.f / (1.f + __expf(-(acc_r[rI] + bir + bhr)));
    const float zg = 1.f / (1.f + __expf(-(acc_z[rI] + biz + bhz)));
    const float ng = tanhf((acc_xn[rI] + bin) + rg * (acc_hn[rI] + bhn));
    const float hv = (1.f - zg) * ng + zg * hprev;
    h_new[(size_t)m * HIDDEN + j] = hv;
    h_out[(size_t)m * HIDDEN + j] = (bf16)hv;
  }
}

// ---------------- output head: [128,512] @ [5,512]^T + b ----------------
__global__ void out_kernel(const float* __restrict__ h, const float* __restrict__ Wout,
                           const float* __restrict__ bout, float* __restrict__ out) {
  const int i = blockIdx.x * blockDim.x + threadIdx.x;
  if (i >= BATCH * NOUT) return;
  const int m = i / NOUT, o = i % NOUT;
  float s = bout[o];
  for (int k = 0; k < HIDDEN; ++k) s += h[(size_t)m * HIDDEN + k] * Wout[(size_t)o * HIDDEN + k];
  out[i] = s;
}

extern "C" void kernel_launch(void* const* d_in, const int* in_sizes, int n_in,
                              void* d_out, int out_size, void* d_ws, size_t ws_size,
                              hipStream_t stream) {
  const int*   x     = (const int*)d_in[0];
  const float* emb   = (const float*)d_in[1];
  const float* W_ih0 = (const float*)d_in[2];
  const float* W_hh0 = (const float*)d_in[3];
  const float* b_ih0 = (const float*)d_in[4];
  const float* b_hh0 = (const float*)d_in[5];
  const float* W_ih1 = (const float*)d_in[6];
  const float* W_hh1 = (const float*)d_in[7];
  const float* b_ih1 = (const float*)d_in[8];
  const float* b_hh1 = (const float*)d_in[9];
  const float* W_out = (const float*)d_in[10];
  const float* b_out = (const float*)d_in[11];
  float* out = (float*)d_out;

  char* ws = (char*)d_ws;
  size_t off = 0;
  auto take = [&](size_t bytes) -> char* {
    char* p = ws + off;
    off += (bytes + 255) & ~(size_t)255;
    return p;
  };

  const size_t NROWS = (size_t)BATCH * TLEN;          // 65536
  bf16*  A0      = (bf16*)take(NROWS * EMBPAD * sizeof(bf16));       // ~40 MB
  bf16*  Wih0b   = (bf16*)take((size_t)GATES * EMBPAD * sizeof(bf16));
  bf16*  Whh0b   = (bf16*)take((size_t)GATES * HIDDEN * sizeof(bf16));
  bf16*  Wih1b   = (bf16*)take((size_t)GATES * HIDDEN * sizeof(bf16));
  bf16*  Whh1b   = (bf16*)take((size_t)GATES * HIDDEN * sizeof(bf16));
  bf16*  hs0     = (bf16*)take(NROWS * HIDDEN * sizeof(bf16));       // ~64 MB, [t][b][k]
  bf16*  zero_bf = (bf16*)take((size_t)BATCH * HIDDEN * sizeof(bf16));
  bf16*  h1bf    = (bf16*)take(2 * (size_t)BATCH * HIDDEN * sizeof(bf16));
  float* h0f     = (float*)take(2 * (size_t)BATCH * HIDDEN * sizeof(float));
  float* h1f     = (float*)take(2 * (size_t)BATCH * HIDDEN * sizeof(float));

  const size_t HB = (size_t)BATCH * HIDDEN;   // 65536 elements

  // init state
  init_zero_kernel<<<(BATCH * HIDDEN + 255) / 256, 256, 0, stream>>>(zero_bf, h1bf, h0f, h1f);

  // embedding gather (time-major, K padded 300 -> 320)
  embed_gather_kernel<<<(int)NROWS, EMBPAD, 0, stream>>>(x, emb, A0);

  // weight converts
  cvt_w_kernel<<<GATES, EMBPAD, 0, stream>>>(W_ih0, Wih0b, EMBDIM, EMBPAD);
  cvt_w_kernel<<<GATES, HIDDEN, 0, stream>>>(W_hh0, Whh0b, HIDDEN, HIDDEN);
  cvt_w_kernel<<<GATES, HIDDEN, 0, stream>>>(W_ih1, Wih1b, HIDDEN, HIDDEN);
  cvt_w_kernel<<<GATES, HIDDEN, 0, stream>>>(W_hh1, Whh1b, HIDDEN, HIDDEN);

  // layer 0 recurrence: writes bf16 h directly into the time-major history hs0
  for (int t = 0; t < TLEN; ++t) {
    const bf16* Ah = (t == 0) ? zero_bf : (hs0 + (size_t)(t - 1) * HB);
    gru_step_kernel<<<32, 256, 0, stream>>>(
        A0 + (size_t)t * BATCH * EMBPAD, EMBPAD, Wih0b,
        Ah, Whh0b, b_ih0, b_hh0,
        h0f + (size_t)(t & 1) * HB,
        h0f + (size_t)((t + 1) & 1) * HB,
        hs0 + (size_t)t * HB);
  }

  // layer 1 recurrence: input is hs0[t]; only final h needed
  for (int t = 0; t < TLEN; ++t) {
    const bf16* Ah = (t == 0) ? zero_bf : (h1bf + (size_t)(t & 1) * HB);
    gru_step_kernel<<<32, 256, 0, stream>>>(
        hs0 + (size_t)t * HB, HIDDEN, Wih1b,
        Ah, Whh1b, b_ih1, b_hh1,
        h1f + (size_t)(t & 1) * HB,
        h1f + (size_t)((t + 1) & 1) * HB,
        h1bf + (size_t)((t + 1) & 1) * HB);
  }

  // final head: h at t=511 lives in parity buffer (512 & 1) == 0
  out_kernel<<<(BATCH * NOUT + 255) / 256, 256, 0, stream>>>(h1f, W_out, b_out, out);
}
